// Decoder_6511170421087
// MI455X (gfx1250) — matrix operand order repaired and checked by reading, not compiled, not run on hardware
//
#include <hip/hip_runtime.h>
#include <hip/hip_bf16.h>

// ---------------- model dims ----------------
#define V_    12000
#define WE_   1000
#define FEAT_ 2048
#define FE_   1024
#define H_    1024
#define AH_   512
#define NREG_ 36
#define B_    128
#define T_    17
#define NSTEP 16
#define KXATT 3072   // 1024 (h_lang) + 1024 (fc_e) + 1000 (emb) + 24 pad
#define G4H   4096

typedef __bf16 bhalf;
typedef __bf16 v16bf __attribute__((ext_vector_type(16)));
typedef float  v8f   __attribute__((ext_vector_type(8)));

union FragU { v16bf v; uint4 q[2]; };

// One pipeline stage: B fragment + 4 A fragments (64 rows of C).
struct Bufs { FragU fb, fa0, fa1, fa2, fa3; };

__device__ __forceinline__ void load_bufs(Bufs& o, const bhalf* __restrict__ brow,
                                          const bhalf* __restrict__ arow,
                                          size_t askip, int k, int K)
{
  o.fb.q[0] = *(const uint4*)(brow + k);
  o.fb.q[1] = *(const uint4*)(brow + k + 16);
  if (k + 256 < K) __builtin_prefetch(brow + k + 256, 0, 3);   // WGP-scope stream prefetch
  const bhalf* a = arow + k;
  o.fa0.q[0] = *(const uint4*)(a);  o.fa0.q[1] = *(const uint4*)(a + 16);  a += askip;
  o.fa1.q[0] = *(const uint4*)(a);  o.fa1.q[1] = *(const uint4*)(a + 16);  a += askip;
  o.fa2.q[0] = *(const uint4*)(a);  o.fa2.q[1] = *(const uint4*)(a + 16);  a += askip;
  o.fa3.q[0] = *(const uint4*)(a);  o.fa3.q[1] = *(const uint4*)(a + 16);
}

__device__ __forceinline__ void do_wmma(v8f acc[4], const Bufs& b)
{
  acc[0] = __builtin_amdgcn_wmma_f32_16x16x32_bf16(false, b.fa0.v, false, b.fb.v, (short)0, acc[0], false, false);
  acc[1] = __builtin_amdgcn_wmma_f32_16x16x32_bf16(false, b.fa1.v, false, b.fb.v, (short)0, acc[1], false, false);
  acc[2] = __builtin_amdgcn_wmma_f32_16x16x32_bf16(false, b.fa2.v, false, b.fb.v, (short)0, acc[2], false, false);
  acc[3] = __builtin_amdgcn_wmma_f32_16x16x32_bf16(false, b.fa3.v, false, b.fb.v, (short)0, acc[3], false, false);
}

// Double-buffered K loop (requires K % 64 == 0; all K here are 1024/2048/3072).
__device__ __forceinline__ void gemm_pass(v8f acc[4], const bhalf* __restrict__ A,
                                          const bhalf* __restrict__ W, int K,
                                          int m0, int n0, int idx, int kbA, int kbB)
{
  const bhalf* brow = W + (size_t)(n0 + idx) * K + kbB;
  const bhalf* arow = A + (size_t)(m0 + idx) * K + kbA;
  const size_t askip = (size_t)16 * K;

  Bufs b0, b1;
  load_bufs(b0, brow, arow, askip, 0, K);
  int k = 0;
  for (; k < K - 64; k += 64) {
    load_bufs(b1, brow, arow, askip, k + 32, K);   // issue next loads first
    do_wmma(acc, b0);                              // overlap WMMA with them
    load_bufs(b0, brow, arow, askip, k + 64, K);
    do_wmma(acc, b1);
  }
  load_bufs(b1, brow, arow, askip, k + 32, K);
  do_wmma(acc, b0);
  do_wmma(acc, b1);
}

// =====================================================================
// Generic dual-source bf16 WMMA GEMM:
//   C[M,N] = A1[M,K1] @ W1[N,K1]^T  (+ A2[M,K2] @ W2[N,K2]^T) + bias1 + bias2
// Block = 8 waves mapped as 2 m-groups x 4 n-tiles -> block covers 128 rows
// x 64 cols, so each weight row is pulled from L2 once per kernel (2x reuse
// inside the block hits L0 of the same WGP). Each wave: 4 accumulator tiles,
// double-buffered fragment pipeline.
// Requires: M % 128 == 0, K % 64 == 0, N % 16 == 0. blockDim = 256.
// =====================================================================
__global__ __launch_bounds__(256) void gemm_wmma_kernel(
    const bhalf* __restrict__ A1, const bhalf* __restrict__ W1, int K1,
    const bhalf* __restrict__ A2, const bhalf* __restrict__ W2, int K2,
    const float* __restrict__ bias1, const float* __restrict__ bias2,
    float* __restrict__ Cf, int ldcf,
    bhalf* __restrict__ Cb, int ldcb,
    bhalf* __restrict__ Cb2, int ldcb2,
    int N, int relu)
{
  const int wave = threadIdx.x >> 5;
  const int lane = threadIdx.x & 31;
  const int half = lane >> 4;
  const int idx  = lane & 15;
  const int ng   = wave & 3;            // n-tile within block
  const int mg   = wave >> 2;           // m-group (0: rows 0-63, 1: rows 64-127)
  const int nt   = blockIdx.x * 4 + ng;
  if (nt * 16 >= N) return;             // whole wave exits together (EXEC stays all-1)
  const int n0  = nt * 16;
  const int m0  = blockIdx.y * 128 + mg * 64;
  const int kbA = half * 8;             // A frag: K = kb..kb+7, kb+16..kb+23 (ISA 16-bit A 16x32)
  const int kbB = half * 8;             // B frag: K = kb..kb+7, kb+16..kb+23 (same as A)

  v8f acc[4];
  const v8f vzero = {0.f,0.f,0.f,0.f,0.f,0.f,0.f,0.f};
#pragma unroll
  for (int mt = 0; mt < 4; ++mt) acc[mt] = vzero;

  gemm_pass(acc, A1, W1, K1, m0, n0, idx, kbA, kbB);
  if (A2) gemm_pass(acc, A2, W2, K2, m0, n0, idx, kbA, kbB);

  // ---- epilogue: bias, relu, stores (f32 and/or bf16) ----
  const int col = n0 + idx;
  float bs = 0.f;
  if (bias1) bs += bias1[col];
  if (bias2) bs += bias2[col];
#pragma unroll
  for (int mt = 0; mt < 4; ++mt) {
#pragma unroll
    for (int r = 0; r < 8; ++r) {
      int row = m0 + mt * 16 + half * 8 + r;  // C/D layout: VGPR r -> M=r (lanes0-15), M=r+8 (16-31)
      float v = acc[mt][r] + bs;
      if (relu) v = fmaxf(v, 0.f);
      if (Cf)  Cf [(size_t)row * ldcf  + col] = v;
      if (Cb)  Cb [(size_t)row * ldcb  + col] = (bhalf)v;
      if (Cb2) Cb2[(size_t)row * ldcb2 + col] = (bhalf)v;
    }
  }
}

// ---------------- fp32 -> bf16 convert (optional K zero-pad) ----------------
__global__ void cvt_pad_kernel(const float* __restrict__ in, bhalf* __restrict__ out,
                               int K, int Kp, long total)
{
  long gid = (long)blockIdx.x * blockDim.x + threadIdx.x;
  if (gid >= total) return;
  long n = gid / Kp;
  int  k = (int)(gid - n * Kp);
  float v = (k < K) ? in[n * K + k] : 0.f;
  out[gid] = (bhalf)v;
}

// ---------------- fp32 [R,C] -> bf16 [C,R] transpose-convert ----------------
__global__ void cvt_T_kernel(const float* __restrict__ in, bhalf* __restrict__ out,
                             int R, int C, long total)
{
  long gid = (long)blockIdx.x * blockDim.x + threadIdx.x;
  if (gid >= total) return;
  long c = gid / R;
  int  r = (int)(gid - c * R);
  out[gid] = (bhalf)in[(long)r * C + c];
}

// -------- embedding gather + ReLU -> x_att[:, 2048:3072] (pad zeroed) --------
__global__ void emb_relu_kernel(const float* __restrict__ emb_w, const int* __restrict__ captions,
                                int t, bhalf* __restrict__ x_att)
{
  int gid = blockIdx.x * blockDim.x + threadIdx.x;   // B_*1024 threads
  int b = gid >> 10;
  int j = gid & 1023;
  int id = captions[b * T_ + t];
  float v = 0.f;
  if (j < WE_) {
    v = emb_w[(long)id * WE_ + j];
    v = v > 0.f ? v : 0.f;
  }
  x_att[(long)b * KXATT + 2048 + j] = (bhalf)v;
}

// ---------------- LSTM cell (in-place c), emits bf16 h to two places ----------------
__global__ void lstm_cell_kernel(const float* __restrict__ gates, float* __restrict__ c,
                                 bhalf* __restrict__ h1, int ld1,
                                 bhalf* __restrict__ h2, int ld2)
{
  int gid = blockIdx.x * blockDim.x + threadIdx.x;   // B_*H_ threads
  int b = gid >> 10;
  int j = gid & 1023;
  const float* g = gates + (long)b * G4H;
  float gi = g[j], gf = g[1024 + j], gg = g[2048 + j], go = g[3072 + j];
  float si = 1.f / (1.f + __expf(-gi));
  float sf = 1.f / (1.f + __expf(-gf));
  float so = 1.f / (1.f + __expf(-go));
  float cn = sf * c[gid] + si * tanhf(gg);
  c[gid] = cn;
  float h = so * tanhf(cn);
  h1[(long)b * ld1 + j] = (bhalf)h;
  h2[(long)b * ld2 + j] = (bhalf)h;
}

// ---- attention: e = tanh(p_att + hq)@alpha_w + b; softmax(36); att_res -> x_lang[:,0:1024] ----
__global__ __launch_bounds__(256) void attention_kernel(
    const float* __restrict__ p_att, const float* __restrict__ hq,
    const float* __restrict__ alpha_w, const float* __restrict__ alpha_b,
    const bhalf* __restrict__ att_e_bf, bhalf* __restrict__ x_lang)
{
  const int b = blockIdx.x;
  const int tid = threadIdx.x;
  const int wave = tid >> 5;
  const int lane = tid & 31;
  __shared__ float e_sm[NREG_];
  __shared__ float a_sm[NREG_];

  for (int n = wave; n < NREG_; n += 8) {
    const float* pr = p_att + ((long)b * NREG_ + n) * AH_;
    const float* hr = hq + (long)b * AH_;
    float s = 0.f;
    for (int d = lane; d < AH_; d += 32)
      s += alpha_w[d] * tanhf(pr[d] + hr[d]);
    for (int off = 16; off > 0; off >>= 1)
      s += __shfl_down(s, off, 32);
    if (lane == 0) e_sm[n] = s + alpha_b[0];
  }
  __syncthreads();
  if (tid == 0) {
    float mx = -3.4e38f;
    for (int n = 0; n < NREG_; ++n) mx = fmaxf(mx, e_sm[n]);
    float sum = 0.f;
    for (int n = 0; n < NREG_; ++n) { a_sm[n] = __expf(e_sm[n] - mx); sum += a_sm[n]; }
    float inv = 1.f / sum;
    for (int n = 0; n < NREG_; ++n) a_sm[n] *= inv;
  }
  __syncthreads();
  for (int d = tid; d < FE_; d += 256) {
    float acc = 0.f;
    const bhalf* ae = att_e_bf + (long)b * NREG_ * FE_ + d;
#pragma unroll 4
    for (int n = 0; n < NREG_; ++n)
      acc += a_sm[n] * (float)ae[(long)n * FE_];
    x_lang[(long)b * 2048 + d] = (bhalf)acc;
  }
}

// ---------------- row-wise log-softmax over V=12000, write out[b][t][:] ----------------
__global__ __launch_bounds__(256) void logsoftmax_kernel(
    const float* __restrict__ logits, float* __restrict__ out, int t)
{
  const int b = blockIdx.x;
  const int tid = threadIdx.x;
  __shared__ float red[256];
  const float* row = logits + (long)b * V_;

  float mx = -3.4e38f;
  for (int v = tid; v < V_; v += 256) mx = fmaxf(mx, row[v]);
  red[tid] = mx; __syncthreads();
  for (int s = 128; s > 0; s >>= 1) { if (tid < s) red[tid] = fmaxf(red[tid], red[tid + s]); __syncthreads(); }
  mx = red[0]; __syncthreads();

  float sum = 0.f;
  for (int v = tid; v < V_; v += 256) sum += __expf(row[v] - mx);
  red[tid] = sum; __syncthreads();
  for (int s = 128; s > 0; s >>= 1) { if (tid < s) red[tid] += red[tid + s]; __syncthreads(); }
  float lse = __logf(red[0]) + mx;

  float* o = out + ((long)b * NSTEP + t) * V_;
  for (int v = tid; v < V_; v += 256) o[v] = row[v] - lse;
}

// =====================================================================
// Host side
// =====================================================================
static inline void launch_gemm(hipStream_t s,
    const bhalf* A1, const bhalf* W1, int K1,
    const bhalf* A2, const bhalf* W2, int K2,
    const float* b1, const float* b2,
    float* Cf, int ldcf, bhalf* Cb, int ldcb, bhalf* Cb2, int ldcb2,
    int M, int N, int relu)
{
  dim3 grid((N / 16 + 3) / 4, M / 128);
  gemm_wmma_kernel<<<grid, 256, 0, s>>>(A1, W1, K1, A2, W2, K2, b1, b2,
                                        Cf, ldcf, Cb, ldcb, Cb2, ldcb2, N, relu);
}

extern "C" void kernel_launch(void* const* d_in, const int* in_sizes, int n_in,
                              void* d_out, int out_size, void* d_ws, size_t ws_size,
                              hipStream_t stream)
{
  const float* fc_feats  = (const float*)d_in[0];
  const float* att_feats = (const float*)d_in[1];
  const int*   captions  = (const int*)  d_in[2];
  const float* emb_w     = (const float*)d_in[3];
  const float* fc_w      = (const float*)d_in[4];
  const float* fc_b      = (const float*)d_in[5];
  const float* atte_w    = (const float*)d_in[6];
  const float* atte_b    = (const float*)d_in[7];
  const float* ctx_w     = (const float*)d_in[8];
  const float* ctx_b     = (const float*)d_in[9];
  const float* attl_wih  = (const float*)d_in[10];
  const float* attl_whh  = (const float*)d_in[11];
  const float* attl_bih  = (const float*)d_in[12];
  const float* attl_bhh  = (const float*)d_in[13];
  const float* h2att_w   = (const float*)d_in[14];
  const float* h2att_b   = (const float*)d_in[15];
  const float* alpha_w   = (const float*)d_in[16];
  const float* alpha_b   = (const float*)d_in[17];
  const float* langl_wih = (const float*)d_in[18];
  const float* langl_whh = (const float*)d_in[19];
  const float* langl_bih = (const float*)d_in[20];
  const float* langl_bhh = (const float*)d_in[21];
  const float* cls_w     = (const float*)d_in[22];
  const float* cls_b     = (const float*)d_in[23];
  float* out = (float*)d_out;

  // ---- workspace carve-out ----
  char* ws = (char*)d_ws;
  size_t off = 0;
  auto carve = [&](size_t bytes) -> char* {
    char* p = ws + off; off += (bytes + 255) & ~(size_t)255; return p;
  };
  bhalf* wih_att_bf  = (bhalf*)carve((size_t)G4H * KXATT * 2);   // [4096,3072]
  bhalf* whh_att_bf  = (bhalf*)carve((size_t)G4H * H_ * 2);      // [4096,1024]
  bhalf* wih_lang_bf = (bhalf*)carve((size_t)G4H * 2048 * 2);    // [4096,2048]
  bhalf* whh_lang_bf = (bhalf*)carve((size_t)G4H * H_ * 2);
  bhalf* cls_wT_bf   = (bhalf*)carve((size_t)V_ * H_ * 2);       // [12000,1024]
  bhalf* fc_wT_bf    = (bhalf*)carve((size_t)FE_ * FEAT_ * 2);   // [1024,2048]
  bhalf* atte_wT_bf  = (bhalf*)carve((size_t)FE_ * FEAT_ * 2);
  bhalf* ctx_wT_bf   = (bhalf*)carve((size_t)AH_ * FE_ * 2);     // [512,1024]
  bhalf* h2att_wT_bf = (bhalf*)carve((size_t)AH_ * H_ * 2);
  bhalf* fcf_bf      = (bhalf*)carve((size_t)B_ * FEAT_ * 2);
  bhalf* attf_bf     = (bhalf*)carve((size_t)B_ * NREG_ * FEAT_ * 2);
  bhalf* att_e_bf    = (bhalf*)carve((size_t)B_ * NREG_ * FE_ * 2);
  float* p_att       = (float*)carve((size_t)B_ * NREG_ * AH_ * 4);
  bhalf* x_att       = (bhalf*)carve((size_t)B_ * KXATT * 2);
  bhalf* x_lang      = (bhalf*)carve((size_t)B_ * 2048 * 2);
  bhalf* h_att_bf    = (bhalf*)carve((size_t)B_ * H_ * 2);
  bhalf* h_lang_bf   = (bhalf*)carve((size_t)B_ * H_ * 2);
  float* c_att       = (float*)carve((size_t)B_ * H_ * 4);
  float* c_lang      = (float*)carve((size_t)B_ * H_ * 4);
  float* gates       = (float*)carve((size_t)B_ * G4H * 4);
  float* hq          = (float*)carve((size_t)B_ * AH_ * 4);
  float* logits      = (float*)carve((size_t)B_ * V_ * 4);
  (void)ws_size; (void)in_sizes; (void)n_in; (void)out_size;

  auto cdiv = [](long a, long b) { return (int)((a + b - 1) / b); };

  // ---- zero-init recurrent state (graph-capturable memsets) ----
  hipMemsetAsync(x_att,     0, (size_t)B_ * KXATT * 2, stream);  // h_lang slot = 0 at t=0
  hipMemsetAsync(x_lang,    0, (size_t)B_ * 2048 * 2, stream);
  hipMemsetAsync(h_att_bf,  0, (size_t)B_ * H_ * 2, stream);
  hipMemsetAsync(h_lang_bf, 0, (size_t)B_ * H_ * 2, stream);
  hipMemsetAsync(c_att,     0, (size_t)B_ * H_ * 4, stream);
  hipMemsetAsync(c_lang,    0, (size_t)B_ * H_ * 4, stream);

  // ---- one-time weight conversion to bf16 (decoder weights stay [N,K]) ----
  {
    long t1 = (long)G4H * KXATT;
    cvt_pad_kernel<<<cdiv(t1, 256), 256, 0, stream>>>(attl_wih, wih_att_bf, 3048, KXATT, t1);
    long t2 = (long)G4H * H_;
    cvt_pad_kernel<<<cdiv(t2, 256), 256, 0, stream>>>(attl_whh, whh_att_bf, H_, H_, t2);
    long t3 = (long)G4H * 2048;
    cvt_pad_kernel<<<cdiv(t3, 256), 256, 0, stream>>>(langl_wih, wih_lang_bf, 2048, 2048, t3);
    cvt_pad_kernel<<<cdiv(t2, 256), 256, 0, stream>>>(langl_whh, whh_lang_bf, H_, H_, t2);
    // [K,N] weights -> transposed [N,K] bf16
    long tc = (long)V_ * H_;
    cvt_T_kernel<<<cdiv(tc, 256), 256, 0, stream>>>(cls_w, cls_wT_bf, H_, V_, tc);
    long tf = (long)FE_ * FEAT_;
    cvt_T_kernel<<<cdiv(tf, 256), 256, 0, stream>>>(fc_w, fc_wT_bf, FEAT_, FE_, tf);
    cvt_T_kernel<<<cdiv(tf, 256), 256, 0, stream>>>(atte_w, atte_wT_bf, FEAT_, FE_, tf);
    long tx = (long)AH_ * FE_;
    cvt_T_kernel<<<cdiv(tx, 256), 256, 0, stream>>>(ctx_w, ctx_wT_bf, FE_, AH_, tx);
    cvt_T_kernel<<<cdiv(tx, 256), 256, 0, stream>>>(h2att_w, h2att_wT_bf, H_, AH_, tx);
    // activations -> bf16
    long ta = (long)B_ * FEAT_;
    cvt_pad_kernel<<<cdiv(ta, 256), 256, 0, stream>>>(fc_feats, fcf_bf, FEAT_, FEAT_, ta);
    long tb = (long)B_ * NREG_ * FEAT_;
    cvt_pad_kernel<<<cdiv(tb, 256), 256, 0, stream>>>(att_feats, attf_bf, FEAT_, FEAT_, tb);
  }

  // ---- feature preprocessing GEMMs ----
  // fc_e = relu(fc_feats @ fc_w + fc_b) -> bf16, written directly into x_att[:,1024:2048]
  launch_gemm(stream, fcf_bf, fc_wT_bf, FEAT_, nullptr, nullptr, 0, fc_b, nullptr,
              nullptr, 0, x_att + 1024, KXATT, nullptr, 0, B_, FE_, 1);
  // att_e = relu(att_feats @ atte_w + atte_b) -> bf16 [B*36, 1024]
  launch_gemm(stream, attf_bf, atte_wT_bf, FEAT_, nullptr, nullptr, 0, atte_b, nullptr,
              nullptr, 0, att_e_bf, FE_, nullptr, 0, B_ * NREG_, FE_, 1);
  // p_att = att_e @ ctx_w + ctx_b -> f32 [B*36, 512]
  launch_gemm(stream, att_e_bf, ctx_wT_bf, FE_, nullptr, nullptr, 0, ctx_b, nullptr,
              p_att, AH_, nullptr, 0, nullptr, 0, B_ * NREG_, AH_, 0);

  // ---- recurrent decode loop ----
  for (int t = 0; t < NSTEP; ++t) {
    // x_att[:,2048:3072] = relu(emb[captions[:,t]]) (zero-padded)
    emb_relu_kernel<<<(B_ * 1024) / 256, 256, 0, stream>>>(emb_w, captions, t, x_att);

    // att-LSTM gates = x_att @ attl_wih^T + h_att @ attl_whh^T + biases
    launch_gemm(stream, x_att, wih_att_bf, KXATT, h_att_bf, whh_att_bf, H_,
                attl_bih, attl_bhh, gates, G4H, nullptr, 0, nullptr, 0, B_, G4H, 0);
    // cell update: writes h_att bf16 (standalone + into x_lang[:,1024:2048])
    lstm_cell_kernel<<<(B_ * H_) / 256, 256, 0, stream>>>(gates, c_att,
                                                          h_att_bf, H_, x_lang + 1024, 2048);
    // hq = h_att @ h2att_w + h2att_b
    launch_gemm(stream, h_att_bf, h2att_wT_bf, H_, nullptr, nullptr, 0, h2att_b, nullptr,
                hq, AH_, nullptr, 0, nullptr, 0, B_, AH_, 0);
    // attention: softmax over 36 regions, att_res -> x_lang[:,0:1024]
    attention_kernel<<<B_, 256, 0, stream>>>(p_att, hq, alpha_w, alpha_b, att_e_bf, x_lang);

    // lang-LSTM gates = x_lang @ langl_wih^T + h_lang @ langl_whh^T + biases
    launch_gemm(stream, x_lang, wih_lang_bf, 2048, h_lang_bf, whh_lang_bf, H_,
                langl_bih, langl_bhh, gates, G4H, nullptr, 0, nullptr, 0, B_, G4H, 0);
    // cell update: writes h_lang bf16 (standalone + into x_att[:,0:1024] for next step)
    lstm_cell_kernel<<<(B_ * H_) / 256, 256, 0, stream>>>(gates, c_lang,
                                                          h_lang_bf, H_, x_att, KXATT);

    // logits = h_lang @ cls_w + cls_b  (N=12000)
    launch_gemm(stream, h_lang_bf, cls_wT_bf, H_, nullptr, nullptr, 0, cls_b, nullptr,
                logits, V_, nullptr, 0, nullptr, 0, B_, V_, 0);
    // log_softmax -> out[b][t][:]
    logsoftmax_kernel<<<B_, 256, 0, stream>>>(logits, out, t);
  }
}
